// VanillaRNN_65249143160920
// MI455X (gfx1250) — matrix operand order repaired and checked by reading, not compiled, not run on hardware
//
#include <hip/hip_runtime.h>

typedef __attribute__((ext_vector_type(16))) _Float16 v16h;
typedef __attribute__((ext_vector_type(2)))  _Float16 v2h;
typedef __attribute__((ext_vector_type(8)))  float    v8f;

// gfx1250 hardware TANH trans op (falls back to ocml if unavailable)
#if __has_builtin(__builtin_amdgcn_tanhf)
#define FAST_TANH(v) __builtin_amdgcn_tanhf(v)
#elif __has_builtin(__builtin_amdgcn_tanh_f32)
#define FAST_TANH(v) __builtin_amdgcn_tanh_f32(v)
#else
#define FAST_TANH(v) tanhf(v)
#endif

namespace {
constexpr int SEQ    = 512;
constexpr int HID    = 256;
constexpr int NCLS   = 10;
constexpr int BATCHN = 2048;
constexpr int BT     = 16;                 // batch columns per workgroup (one N-tile)
constexpr int KA     = HID / 32 + 1;       // 9 K-chunks: 8 of W_hh + 1 augmented [W_hx | b_h | 0..]
constexpr int MTILES = HID / 16;           // 16 M-tiles
constexpr int MPW    = 2;                  // M-tiles per wave (8 waves x 2 = 16)
constexpr int FRAG   = 512;                // halves per fragment (32 lanes * 16 halves = 1 KiB)
constexpr int WA_H   = MTILES * KA * FRAG; // 73728 halves = 144 KiB (staging for A-frags)
constexpr int HB_H   = KA * FRAG;          // 4608 halves = 9 KiB per h buffer
constexpr size_t SMEM_BYTES = size_t(WA_H) * 2 + size_t(2 * HB_H) * 2; // ~162 KiB < 320 KiB
}

__global__ __launch_bounds__(256)
void rnn_wmma_f16(const float* __restrict__ x,
                  const float* __restrict__ Whx,
                  const float* __restrict__ Whh,
                  const float* __restrict__ bh,
                  const float* __restrict__ Wph,
                  const float* __restrict__ bp,
                  float* __restrict__ out)
{
    extern __shared__ __align__(16) char smem_raw[];
    _Float16* wA = reinterpret_cast<_Float16*>(smem_raw); // [MTILES*KA][FRAG] A-frag staging
    _Float16* hb = wA + WA_H;                             // [2][KA][FRAG]    h B-fragments

    const int tid  = threadIdx.x;
    const int lane = tid & 31;
    const int wave = tid >> 5;
    const int wg   = blockIdx.x;

    // ---- stage augmented weights [W_hh | W_hx | b_h] into LDS as WMMA A-fragments ----
    // A 16x32 f16 fragment as staged here: lane = M + 16*(K half); dword v -> K 16*half + 2v, +1,
    // i.e. contiguous K 0..15 / 16..31 per lane half (2 halves per dword; same order as the h B-frags).
    for (int idx = tid; idx < WA_H; idx += 256) {
        int frag = idx >> 9;            // mt*KA + kc
        int ls   = (idx >> 4) & 31;     // lane slot
        int hfi  = idx & 15;            // v*2 + half
        int v    = hfi >> 1, hlf = hfi & 1;
        int mt   = frag / KA, kc = frag % KA;
        int kb   = (ls >> 4) * 8;
        int kl   = 2 * kb + hfi;
        int m    = mt * 16 + (ls & 15);
        float val;
        if (kc < KA - 1) val = Whh[m * HID + kc * 32 + kl];                    // W_hh block
        else             val = (kl == 0) ? Whx[m] : (kl == 1) ? bh[m] : 0.0f;  // augmented cols
        wA[idx] = (_Float16)val;
    }
    // h0 = 0 in both ping-pong buffers (zeroes aug rows >= 258 once and for all)
    for (int idx = tid; idx < 2 * HB_H; idx += 256) hb[idx] = (_Float16)0.0f;
    __syncthreads();

    // ---- wave work assignment: 8 waves x 2 M-tiles, single 16-column N-tile ----
    const int mbase = wave * MPW;
    const int nl    = lane & 15;
    const int hi    = lane >> 4;
    const int bcol  = wg * BT + nl;
    const float* xrow = x + (size_t)bcol * SEQ;
    const bool  writer = (wave == 0) && (hi == 0); // 16 lanes own the augmented rows

    // seed augmented rows of buffer 0 with [x_0, 1]
    if (writer) {
        float x0 = xrow[0];
        *reinterpret_cast<v2h*>(&hb[(KA - 1) * FRAG + nl * 16]) =
            (v2h){(_Float16)x0, (_Float16)1.0f};
    }

    // ---- pull this wave's A-fragments into registers for the whole recurrence ----
    // 2 M-tiles x 9 K-chunks x 8 VGPRs = 144 VGPRs, resident across all 512 steps.
    v16h Areg[MPW][KA];
    #pragma unroll
    for (int i = 0; i < MPW; ++i)
        #pragma unroll
        for (int kc = 0; kc < KA; ++kc)
            Areg[i][kc] = *reinterpret_cast<const v16h*>(
                &wA[((mbase + i) * KA + kc) * FRAG + lane * 16]);
    __syncthreads();

    // ---- recurrence: h_{t+1} = tanh(W_aug @ [h_t; x_t; 1]) ----
    for (int t = 0; t < SEQ; ++t) {
        const _Float16* hrd = hb + (t & 1) * HB_H;
        _Float16*       hwr = hb + ((t & 1) ^ 1) * HB_H;

        // issue next-step x load early; latency hides behind the 18 WMMAs below
        float xnext = 0.0f;
        if (writer && t + 1 < SEQ) {
            xnext = xrow[t + 1];
            if (t + 2 < SEQ) __builtin_prefetch(&xrow[t + 2], 0, 0); // global_prefetch_b8
        }

        v8f acc[MPW];
        #pragma unroll
        for (int i = 0; i < MPW; ++i) acc[i] = (v8f){0.f,0.f,0.f,0.f,0.f,0.f,0.f,0.f};

        #pragma unroll
        for (int kc = 0; kc < KA; ++kc) {
            v16h bf = *reinterpret_cast<const v16h*>(&hrd[kc * FRAG + lane * 16]);
            acc[0] = __builtin_amdgcn_wmma_f32_16x16x32_f16(
                false, Areg[0][kc], false, bf, (short)0, acc[0], false, false);
            acc[1] = __builtin_amdgcn_wmma_f32_16x16x32_f16(
                false, Areg[1][kc], false, bf, (short)0, acc[1], false, false);
        }

        // epilogue: h_new = tanh(acc), packed back into B-fragment layout (f16 pairs)
        #pragma unroll
        for (int i = 0; i < MPW; ++i) {
            int mt = mbase + i;
            #pragma unroll
            for (int r = 0; r < 8; r += 2) {
                int m0 = mt * 16 + hi * 8 + r;          // C/D layout: vgpr r <-> M = r + 8*hi
                float v0 = FAST_TANH(acc[i][r]);
                float v1 = FAST_TANH(acc[i][r + 1]);
                int g     = m0 >> 5;                    // B-frag K-chunk
                int lslot = nl | (((m0 >> 4) & 1) << 4);
                int dpos  = (m0 & 15) >> 1;
                *reinterpret_cast<v2h*>(&hwr[g * FRAG + lslot * 16 + dpos * 2]) =
                    (v2h){(_Float16)v0, (_Float16)v1};
            }
        }
        // feed x_{t+1} (and the bias 1) into the augmented rows of the next buffer
        if (writer && t + 1 < SEQ) {
            *reinterpret_cast<v2h*>(&hwr[(KA - 1) * FRAG + nl * 16]) =
                (v2h){(_Float16)xnext, (_Float16)1.0f};
        }
        __syncthreads();
    }

    // ---- projection: p = W_ph @ h_T + b_p -> out[B, C]; final h is in buffer 0 (SEQ even) ----
    const _Float16* hfin = hb;
    for (int j = tid; j < BT * NCLS; j += 256) {
        int n = j / NCLS;
        int c = j % NCLS;
        float sum = bp[c];
        for (int m = 0; m < HID; ++m) {
            int lslot = n | (((m >> 4) & 1) << 4);
            float hv  = (float)hfin[(m >> 5) * FRAG + lslot * 16 + ((m & 15) >> 1) * 2 + (m & 1)];
            sum += Wph[c * HID + m] * hv;
        }
        out[(size_t)(wg * BT + n) * NCLS + c] = sum;
    }
}

extern "C" void kernel_launch(void* const* d_in, const int* in_sizes, int n_in,
                              void* d_out, int out_size, void* d_ws, size_t ws_size,
                              hipStream_t stream) {
    const float* x   = (const float*)d_in[0];
    const float* Whx = (const float*)d_in[1];
    const float* Whh = (const float*)d_in[2];
    const float* bh  = (const float*)d_in[3];
    const float* Wph = (const float*)d_in[4];
    const float* bp  = (const float*)d_in[5];
    float* out = (float*)d_out;
    (void)in_sizes; (void)n_in; (void)out_size; (void)d_ws; (void)ws_size;

    (void)hipFuncSetAttribute((const void*)rnn_wmma_f16,
                              hipFuncAttributeMaxDynamicSharedMemorySize,
                              (int)SMEM_BYTES);

    dim3 grid(BATCHN / BT);   // 128 workgroups, 16 batch columns each
    dim3 block(256);          // 8 wave32 waves
    rnn_wmma_f16<<<grid, block, SMEM_BYTES, stream>>>(x, Whx, Whh, bh, Wph, bp, out);
}
